// IlluminationPeakLayer_19370302505466
// MI455X (gfx1250) — matrix operand order repaired and checked by reading, not compiled, not run on hardware
//
#include <hip/hip_runtime.h>
#include <hip/hip_bf16.h>
#include <math.h>

// CDNA5 WMMA vector types
typedef __attribute__((ext_vector_type(16))) __bf16 v16bf;
typedef __attribute__((ext_vector_type(8)))  float  v8f;

#define NT     1024      // time bins
#define KOUT   8         // output channels
#define NBATCH 32768
#define THREADS 256                 // 8 waves (wave32)
#define BATCH_PER_WAVE 16
#define BATCH_PER_BLOCK 128         // 8 waves * 16 batches

// ---------------------------------------------------------------------------
// Kernel 1 (tiny, one block):
//   inp[t]   = sum_s relu(illum[s]) * irf[(t+511-s) & 1023]
//   area     = sum_t inp[t]
//   conv2[t] = sum_s inp[s] * irf[(t+511-s) & 1023]
// Because the reference clamp min(inp*scale, 5*pc) provably never binds
// (inp[s] <= area  =>  inp[s]*pc/area <= pc < 5*pc), the per-batch conv
// factorizes to scale_b * conv2 — so conv2 is all the main kernel needs.
// ---------------------------------------------------------------------------
__global__ void __launch_bounds__(256) prep_kernel(const float* __restrict__ illum,
                                                   const float* __restrict__ irf,
                                                   float* __restrict__ ws) {
    __shared__ float s_x[NT];     // relu(illum)
    __shared__ float s_y[NT];     // inp
    __shared__ float s_irf[NT];
    __shared__ float s_red[8];
    const int tid = threadIdx.x;
    for (int i = tid; i < NT; i += 256) {
        s_x[i]   = fmaxf(illum[i], 0.f);
        s_irf[i] = irf[i];
    }
    __syncthreads();
    float lsum = 0.f;
    for (int t = tid; t < NT; t += 256) {
        float acc = 0.f;
        #pragma unroll 4
        for (int s = 0; s < NT; ++s)
            acc = fmaf(s_x[s], s_irf[(t + 511 - s) & (NT - 1)], acc);
        s_y[t] = acc;
        lsum += acc;
    }
    for (int off = 16; off; off >>= 1) lsum += __shfl_xor(lsum, off, 32);
    if ((tid & 31) == 0) s_red[tid >> 5] = lsum;
    __syncthreads();                       // covers s_y writes and s_red writes
    if (tid == 0) {
        float a = 0.f;
        for (int w = 0; w < 8; ++w) a += s_red[w];
        ws[NT] = a;                        // area
    }
    for (int t = tid; t < NT; t += 256) {
        float acc = 0.f;
        #pragma unroll 4
        for (int s = 0; s < NT; ++s)
            acc = fmaf(s_y[s], s_irf[(t + 511 - s) & (NT - 1)], acc);
        ws[t] = acc;                       // conv2
    }
}

// ---------------------------------------------------------------------------
// Counter-based Poisson sampler (deterministic in (batch,bin))
// ---------------------------------------------------------------------------
__device__ __forceinline__ uint32_t hash32(uint32_t x) {
    x ^= x >> 16; x *= 0x7feb352du;
    x ^= x >> 15; x *= 0x846ca68bu;
    x ^= x >> 16; return x;
}

__device__ __forceinline__ float poisson_sample(uint32_t seed, float rate) {
    if (!(rate > 0.f)) return 0.f;
    uint32_t s = hash32(seed * 0x9E3779B9u + 42u);
    if (rate < 12.f) {
        // Knuth product method (expected iters = rate+1; mean rate here ~2)
        const float L = __expf(-rate);
        float p = 1.f;
        int   k = 0;
        do {
            s = hash32(s + 0x9E3779B9u);
            float u = ((float)(s >> 8) + 0.5f) * (1.f / 16777216.f);
            p *= u;
            ++k;
        } while (p > L && k < 64);
        return (float)(k - 1);
    }
    // normal approximation via Box-Muller (rate in [12, ~22])
    uint32_t s2 = hash32(s ^ 0x85ebca6bu);
    float u1 = ((float)(s  >> 8) + 0.5f) * (1.f / 16777216.f);
    float u2 = ((float)(s2 >> 8) + 0.5f) * (1.f / 16777216.f);
    float z  = __fsqrt_rn(-2.f * __logf(u1)) * __cosf(6.2831853071f * u2);
    return fmaxf(0.f, floorf(rate + __fsqrt_rn(rate) * z + 0.5f));
}

// ---------------------------------------------------------------------------
// Kernel 2 (fused): per (b,t) rate -> Poisson -> WMMA projection onto W^T.
//   rate[b,t] = scale_b * conv2[(t - shift_b) & 1023] + amb_b
//   out[b,k]  = sum_t Poisson(rate[b,t]) * W[k,t]
// The projection is a [16b x 32t]x[32t x 16k] WMMA chain (K=1024, 32 steps);
// the C accumulator IS the output accumulator. A-fragments are built in
// t-space so the per-batch circular shift is just an index offset into a
// doubled conv2 array, keeping the B operand (W^T, bf16 in LDS) row-shared.
// ---------------------------------------------------------------------------
__global__ void __launch_bounds__(THREADS) fused_kernel(
        const float* __restrict__ pc,   const float* __restrict__ sbr,
        const float* __restrict__ W,    const int*   __restrict__ bins,
        const float* __restrict__ ws,   float* __restrict__ out) {
    __shared__ float  s_d2[2 * NT];                       // doubled conv2, 8 KB
    __shared__ __align__(16) __bf16 s_wtb[16 * NT];       // W^T bf16 [k][t], 32 KB (k>=8 zero)
    __shared__ float s_scale[BATCH_PER_BLOCK];
    __shared__ float s_amb[BATCH_PER_BLOCK];
    __shared__ int   s_shift[BATCH_PER_BLOCK];

    const int tid = threadIdx.x;
    for (int i = tid; i < 2 * NT; i += THREADS)
        s_d2[i] = ws[i & (NT - 1)];
    for (int i = tid; i < 16 * NT; i += THREADS) {
        const int k = i >> 10;                            // W is [KOUT][NT] row-major
        s_wtb[i] = (k < KOUT) ? (__bf16)W[i] : (__bf16)0.f;
    }
    if (tid < BATCH_PER_BLOCK) {
        const float area = ws[NT];
        const int   gb   = blockIdx.x * BATCH_PER_BLOCK + tid;
        const float p    = pc[gb];
        s_scale[tid] = p / area;
        s_amb[tid]   = (p / sbr[gb]) * (1.f / (float)NT);
        s_shift[tid] = bins[gb] & (NT - 1);
    }
    __syncthreads();

    const int lane = tid & 31;
    const int wave = tid >> 5;
    const int col  = lane & 15;
    const int hi   = lane >> 4;

    // A-operand row M = lane%16 -> this lane's batch
    const int lb = wave * BATCH_PER_WAVE + col;
    const int gb = blockIdx.x * BATCH_PER_BLOCK + lb;
    const float    sc   = s_scale[lb];
    const float    amb  = s_amb[lb];
    const int      base = NT - s_shift[lb];               // doubled-array offset
    const uint32_t seed0 = (uint32_t)gb * (uint32_t)NT;

    v8f acc = {0.f, 0.f, 0.f, 0.f, 0.f, 0.f, 0.f, 0.f};

    for (int ks = 0; ks < NT / 32; ++ks) {
        const int kbase = ks * 32;
        // --- A fragment: Poisson counts in t-space ---
        // A 16-bit layout: t = kbase + 16*hi + h (matches B)
        v16bf a;
        const int t0 = kbase + 16 * hi;
        #pragma unroll
        for (int h = 0; h < 8; ++h) {
            const int   t    = t0 + h;
            const float rate = fmaxf(fmaf(sc, s_d2[base + t], amb), 0.f);
            a[h] = (__bf16)poisson_sample(seed0 + (uint32_t)t, rate);
        }
        const int t1 = kbase + 16 * hi + 8;
        #pragma unroll
        for (int h = 0; h < 8; ++h) {
            const int   t    = t1 + h;
            const float rate = fmaxf(fmaf(sc, s_d2[base + t], amb), 0.f);
            a[8 + h] = (__bf16)poisson_sample(seed0 + (uint32_t)t, rate);
        }
        // --- B fragment: W^T, 32 contiguous bytes -> 2 x ds_load_b128 ---
        // B: N = lane%16 (=k); contiguous K order used here (same as A), K = 16*hi + h  -> t = kbase + 16*hi + h
        union { uint4 q[2]; v16bf v; } bw;
        const uint4* p = (const uint4*)&s_wtb[col * NT + kbase + 16 * hi];
        bw.q[0] = p[0];
        bw.q[1] = p[1];

        acc = __builtin_amdgcn_wmma_f32_16x16x32_bf16(false, a, false, bw.v,
                                                      (short)0, acc, false, false);
    }

    // C layout: lane holds out[b = btile + r + 8*hi, k = col]; cols 8..15 are padding
    if (col < KOUT) {
        const int btile = blockIdx.x * BATCH_PER_BLOCK + wave * BATCH_PER_WAVE;
        #pragma unroll
        for (int r = 0; r < 8; ++r) {
            const int b = btile + r + 8 * hi;
            out[(size_t)b * KOUT + col] = acc[r];
        }
    }
}

// ---------------------------------------------------------------------------
extern "C" void kernel_launch(void* const* d_in, const int* in_sizes, int n_in,
                              void* d_out, int out_size, void* d_ws, size_t ws_size,
                              hipStream_t stream) {
    (void)in_sizes; (void)n_in; (void)out_size; (void)ws_size;
    const float* illum = (const float*)d_in[0];
    const float* pc    = (const float*)d_in[1];
    const float* sbr   = (const float*)d_in[2];
    const float* irf   = (const float*)d_in[3];
    const float* W     = (const float*)d_in[4];   // [KOUT, NT]
    const int*   bins  = (const int*)d_in[5];
    float* ws  = (float*)d_ws;                    // [0..NT) = conv2, [NT] = area
    float* out = (float*)d_out;                   // [NBATCH, KOUT]

    prep_kernel<<<1, 256, 0, stream>>>(illum, irf, ws);
    fused_kernel<<<NBATCH / BATCH_PER_BLOCK, THREADS, 0, stream>>>(
        pc, sbr, W, bins, ws, out);
}
